// SpatialRefine_38955353374756
// MI455X (gfx1250) — matrix operand order repaired and checked by reading, not compiled, not run on hardware
//
#include <hip/hip_runtime.h>
#include <hip/hip_bf16.h>

#define EPS 1e-5f

typedef __attribute__((ext_vector_type(16))) __bf16 v16bf;
typedef __attribute__((ext_vector_type(8)))  float  v8f;
typedef __attribute__((ext_vector_type(4)))  float  f4;

union BF16x16 { v16bf v; f4 f[2]; };

// Single-instruction relu.  Every IR-level spelling (fmaxf / select / fmed3)
// gets canonicalized by LLVM into maxnum(maxnum(x,x), 0) = 2 VALU ops, so we
// pin the exact hardware op: v_max_num_f32 dst, 0, src (inline const in src0).
__device__ __forceinline__ float relu1(float x) {
    float r;
    asm("v_max_num_f32 %0, 0, %1" : "=v"(r) : "v"(x));
    return r;
}

// ---------------------------------------------------------------------------
// Kernel A: 1x1 conv (W[64,256] x fea[n,256,4096]) + per-column L2 normalize,
// storing f^T as bf16 in ft[n][p][k]  (k contiguous, 64 per row = 128B rows).
// Grid: (4096/64, 8), block 256.
// ---------------------------------------------------------------------------
__global__ __launch_bounds__(256)
void conv_norm_kernel(const float* __restrict__ fea,
                      const float* __restrict__ W,
                      __hip_bfloat16* __restrict__ ft) {
    __shared__ float Wlds[256 * 64];     // W transposed: [c][k], 64 KB
    __shared__ float red[64][4];

    const int tid = threadIdx.x;
    const int n   = blockIdx.y;
    const int p0  = blockIdx.x * 64;

    for (int i = tid; i < 64 * 256; i += 256) {
        const int k = i >> 8;
        const int c = i & 255;
        Wlds[c * 64 + k] = W[i];
    }
    __syncthreads();

    const int pl = tid & 63;
    const int kg = tid >> 6;
    const int p  = p0 + pl;

    float acc[16];
#pragma unroll
    for (int j = 0; j < 16; ++j) acc[j] = 0.0f;

    const float* feap = fea + (size_t)n * 256 * 4096 + p;
    for (int c = 0; c < 256; ++c) {
        const float v = feap[(size_t)c * 4096];
        const float* wr = &Wlds[c * 64 + kg * 16];
#pragma unroll
        for (int j4 = 0; j4 < 4; ++j4) {
            const f4 w = *(const f4*)(wr + j4 * 4);
#pragma unroll
            for (int e = 0; e < 4; ++e) acc[j4 * 4 + e] += w[e] * v;
        }
    }

    float ss = 0.0f;
#pragma unroll
    for (int j = 0; j < 16; ++j) ss += acc[j] * acc[j];
    red[pl][kg] = ss;
    __syncthreads();
    const float tot = red[pl][0] + red[pl][1] + red[pl][2] + red[pl][3];
    const float scale = 1.0f / (sqrtf(tot) + EPS);

    __hip_bfloat16* dst = ft + ((size_t)(n * 4096 + p)) * 64 + kg * 16;
#pragma unroll
    for (int j = 0; j < 16; ++j) dst[j] = __float2bfloat16(acc[j] * scale);
}

// ---------------------------------------------------------------------------
// Kernel B: fused  aff = relu(f^T f), column-sum normalize, pred aggregation.
// out[n,c,q] = sum_p pred[n,c,p]*relu(s_pq) / (sum_p relu(s_pq) + EPS)
// One wave owns TWO 16-wide q tiles (32 q): each streamed A tile + pred tile
// feeds 4 accumulated v_wmma_f32_16x16x32_bf16 ops.  Parallel per-row
// accumulators avoid serialized add chains.  Grid: (4096/128, 8), block 128.
// ---------------------------------------------------------------------------
__global__ __launch_bounds__(128)
void affinity_agg_kernel(const float* __restrict__ pred,
                         const __bf16* __restrict__ ft,
                         float* __restrict__ out) {
    __shared__ float plds[2 * 4096];     // pred for this batch, 32 KB

    const int tid = threadIdx.x;
    const int n   = blockIdx.y;

    for (int i = tid; i < 8192; i += 128) plds[i] = pred[(size_t)n * 8192 + i];
    __syncthreads();

    const int lane  = tid & 31;
    const int wave  = tid >> 5;
    const int l16   = lane & 15;
    const int hi    = lane >> 4;             // 0: lanes 0-15, 1: lanes 16-31
    const int qbase = blockIdx.x * 128 + wave * 32;
    const int q0    = qbase + l16;           // tile 0 column
    const int q1    = qbase + 16 + l16;      // tile 1 column

    const __bf16* ftn = ft + (size_t)n * 4096 * 64;

    // B tiles (32x16, K x N): loop invariant.  Lane = column N; lanes 0-15
    // carry K=0..15, lanes 16-31 carry K=16..31 (per K-chunk).
    BF16x16 b00, b01, b10, b11;
    {
        const __bf16* br0 = ftn + (size_t)q0 * 64;
        b00.f[0] = *(const f4*)(br0 + hi * 16);
        b00.f[1] = *(const f4*)(br0 + hi * 16 + 8);
        b01.f[0] = *(const f4*)(br0 + 32 + hi * 16);
        b01.f[1] = *(const f4*)(br0 + 32 + hi * 16 + 8);
        const __bf16* br1 = ftn + (size_t)q1 * 64;
        b10.f[0] = *(const f4*)(br1 + hi * 16);
        b10.f[1] = *(const f4*)(br1 + hi * 16 + 8);
        b11.f[0] = *(const f4*)(br1 + 32 + hi * 16);
        b11.f[1] = *(const f4*)(br1 + 32 + hi * 16 + 8);
    }

    // Parallel accumulators: index r corresponds to C-layout VGPR r
    // (row p = p0 + hi*8 + r for this lane).
    float den0[8], n00[8], n01[8], den1[8], n10[8], n11[8];
#pragma unroll
    for (int r = 0; r < 8; ++r) {
        den0[r] = 0.f; n00[r] = 0.f; n01[r] = 0.f;
        den1[r] = 0.f; n10[r] = 0.f; n11[r] = 0.f;
    }

    for (int p0 = 0; p0 < 4096; p0 += 16) {
        // A tile (16x32, M x K): lane holds row M = l16.
        // lanes 0-15: K=0-15 ; lanes 16-31: K=16-31.
        const __bf16* arow = ftn + (size_t)(p0 + l16) * 64;
        BF16x16 a0, a1;
        a0.f[0] = *(const f4*)(arow + hi * 16);
        a0.f[1] = *(const f4*)(arow + hi * 16 + 8);
        a1.f[0] = *(const f4*)(arow + 32 + hi * 16);
        a1.f[1] = *(const f4*)(arow + 32 + hi * 16 + 8);

        if (p0 + 16 < 4096)
            __builtin_prefetch(arow + 16 * 64, 0, 3);   // global_prefetch next A tile

        v8f c0 = {}, c1 = {};
        c0 = __builtin_amdgcn_wmma_f32_16x16x32_bf16(false, a0.v, false, b00.v,
                                                     (short)0, c0, false, false);
        c0 = __builtin_amdgcn_wmma_f32_16x16x32_bf16(false, a1.v, false, b01.v,
                                                     (short)0, c0, false, false);
        c1 = __builtin_amdgcn_wmma_f32_16x16x32_bf16(false, a0.v, false, b10.v,
                                                     (short)0, c1, false, false);
        c1 = __builtin_amdgcn_wmma_f32_16x16x32_bf16(false, a1.v, false, b11.v,
                                                     (short)0, c1, false, false);

        // pred rows for this lane: p = p0 + hi*8 + r, contiguous 8 floats.
        float pr0[8], pr1[8];
        *(f4*)&pr0[0] = *(const f4*)&plds[p0 + hi * 8];
        *(f4*)&pr0[4] = *(const f4*)&plds[p0 + hi * 8 + 4];
        *(f4*)&pr1[0] = *(const f4*)&plds[4096 + p0 + hi * 8];
        *(f4*)&pr1[4] = *(const f4*)&plds[4096 + p0 + hi * 8 + 4];

#pragma unroll
        for (int r = 0; r < 8; ++r) {
            const float s0 = relu1(c0[r]);   // single v_max_num_f32
            const float s1 = relu1(c1[r]);
            den0[r] += s0;            den1[r] += s1;
            n00[r]  += s0 * pr0[r];   n01[r]  += s0 * pr1[r];
            n10[r]  += s1 * pr0[r];   n11[r]  += s1 * pr1[r];
        }
    }

    // Reduce the 8 parallel accumulators.
    float d0 = 0.f, a00 = 0.f, a01 = 0.f, d1 = 0.f, a10 = 0.f, a11 = 0.f;
#pragma unroll
    for (int r = 0; r < 8; ++r) {
        d0 += den0[r]; a00 += n00[r]; a01 += n01[r];
        d1 += den1[r]; a10 += n10[r]; a11 += n11[r];
    }

    // Each q appears in lanes l16 and l16+16 (p rows 0-7 / 8-15 of each tile).
    d0  += __shfl_xor(d0,  16, 32);
    a00 += __shfl_xor(a00, 16, 32);
    a01 += __shfl_xor(a01, 16, 32);
    d1  += __shfl_xor(d1,  16, 32);
    a10 += __shfl_xor(a10, 16, 32);
    a11 += __shfl_xor(a11, 16, 32);

    if (hi == 0) {
        const float inv0 = 1.0f / (d0 + EPS);
        const float inv1 = 1.0f / (d1 + EPS);
        out[(size_t)n * 8192 + q0]        = a00 * inv0;
        out[(size_t)n * 8192 + 4096 + q0] = a01 * inv0;
        out[(size_t)n * 8192 + q1]        = a10 * inv1;
        out[(size_t)n * 8192 + 4096 + q1] = a11 * inv1;
    }
}

extern "C" void kernel_launch(void* const* d_in, const int* in_sizes, int n_in,
                              void* d_out, int out_size, void* d_ws, size_t ws_size,
                              hipStream_t stream) {
    const float* pred = (const float*)d_in[0];   // (8, 2, 64, 64)
    const float* fea  = (const float*)d_in[1];   // (8, 256, 64, 64)
    const float* W    = (const float*)d_in[2];   // (64, 256)
    float* out = (float*)d_out;                  // (8, 2, 64, 64)

    // Workspace: normalized features f^T as bf16, [8][4096][64] = 4 MB.
    __hip_bfloat16* ft = (__hip_bfloat16*)d_ws;

    dim3 gA(4096 / 64, 8);
    conv_norm_kernel<<<gA, 256, 0, stream>>>(fea, W, ft);

    dim3 gB(4096 / 128, 8);
    affinity_agg_kernel<<<gB, 128, 0, stream>>>(pred, (const __bf16*)ft, out);
}
